// people_local_fusion_48223892799591
// MI455X (gfx1250) — matrix operand order repaired and checked by reading, not compiled, not run on hardware
//
#include <hip/hip_runtime.h>
#include <hip/hip_bf16.h>

typedef __attribute__((ext_vector_type(16))) __bf16 v16bf;
typedef __attribute__((ext_vector_type(8)))  __bf16 v8bf;
typedef __attribute__((ext_vector_type(4)))  __bf16 v4bf;
typedef __attribute__((ext_vector_type(8)))  float  v8f;

#define NIMG 16
#define C    768
#define HW   64
#define SPAT 4096               // 64*64
#define INNER 1024
#define GELEM 98304             // 24*4096 (one group, contiguous in NCHW)
#define EPS   1e-5f

static __device__ __forceinline__ v16bf join16(v8bf lo, v8bf hi) {
  return __builtin_shufflevector(lo, hi, 0,1,2,3,4,5,6,7,8,9,10,11,12,13,14,15);
}

// gfx1250 async global->LDS copy (ASYNCcnt-tracked). LDS byte offset obtained by
// truncating the flat pointer to its low 32 bits (ISA 10.2 aperture rule).
static __device__ __forceinline__ void async_ld_b128(unsigned lds_off, const void* g) {
  asm volatile("global_load_async_to_lds_b128 %0, %1, off"
               :: "v"(lds_off), "v"((unsigned long long)(size_t)g)
               : "memory");
}
static __device__ __forceinline__ void wait_async0() {
  asm volatile("s_wait_asynccnt 0x0" ::: "memory");
}

// ---------------------------------------------------------------- GroupNorm stats
__global__ __launch_bounds__(256)
void gn_stats_k(const float* __restrict__ x, float* __restrict__ stats) {
  const int grp = blockIdx.x;                 // n*32+g, 512 total
  const long base = (long)grp * GELEM;
  __shared__ float r1[256], r2[256];
  float s = 0.f, s2 = 0.f;
  for (int i = threadIdx.x; i < GELEM; i += 256) {
    float v = x[base + i]; s += v; s2 += v * v;
  }
  r1[threadIdx.x] = s; r2[threadIdx.x] = s2; __syncthreads();
  for (int off = 128; off > 0; off >>= 1) {
    if (threadIdx.x < off) { r1[threadIdx.x] += r1[threadIdx.x+off]; r2[threadIdx.x] += r2[threadIdx.x+off]; }
    __syncthreads();
  }
  if (threadIdx.x == 0) {
    float mean = r1[0] / (float)GELEM;
    float var  = r2[0] / (float)GELEM - mean * mean;
    stats[grp * 2 + 0] = mean;
    stats[grp * 2 + 1] = rsqrtf(var + EPS);
  }
}

// --------------------------------------------------- GN apply + SiLU -> bf16
__global__ __launch_bounds__(256)
void gn_silu_k(const float* __restrict__ x, const float* __restrict__ stats,
               const float* __restrict__ gg, const float* __restrict__ gb,
               __bf16* __restrict__ y) {
  const long i = ((long)blockIdx.x * 256 + threadIdx.x) * 4;
  const int c   = (int)((i >> 12) % C);
  const int grp = (int)(i / GELEM);
  const float mean = stats[grp * 2], inv = stats[grp * 2 + 1];
  const float a = gg[c] * inv;
  const float b = gb[c] - mean * a;
  float4 v = *(const float4*)&x[i];
  float o0 = v.x * a + b, o1 = v.y * a + b, o2 = v.z * a + b, o3 = v.w * a + b;
  o0 = o0 / (1.f + __expf(-o0));
  o1 = o1 / (1.f + __expf(-o1));
  o2 = o2 / (1.f + __expf(-o2));
  o3 = o3 / (1.f + __expf(-o3));
  v4bf out; out[0]=(__bf16)o0; out[1]=(__bf16)o1; out[2]=(__bf16)o2; out[3]=(__bf16)o3;
  *(v4bf*)&y[i] = out;
}

// ---------------------------------------------- conv weight repack: [tap][cout][cin] bf16
__global__ __launch_bounds__(256)
void wpack_k(const float* __restrict__ w, __bf16* __restrict__ wp) {
  const long o = (long)blockIdx.x * 256 + threadIdx.x;        // 14745600 total
  const int cin  = (int)(o % C);
  const int cout = (int)((o / C) % C);
  const int tap  = (int)(o / ((long)C * C));
  wp[o] = (__bf16)w[((long)cout * C + cin) * 25 + tap];
}

// ------------------------------------------------------ 5x5 conv as implicit WMMA GEMM
// grid: (64 rows, 6 cout-blocks of 128, 16 images); block = 256 (8 waves).
// Workgroup computes hmap[img, M0:M0+128, row, 0:64].
// K loop = 5 ky x 24 cin-chunks; each iteration stages all 5 kx weight tiles (async
// global->LDS) plus one halo-extended activation row tile, then runs 20 WMMAs.
// Double-buffered LDS software pipeline overlaps async copies with WMMA issue.
#define ATILE (5 * 128 * 32)     // bf16 elems per A buffer (5 taps)
#define BTILE (68 * 32)          // bf16 elems per B buffer (halo row)

__global__ __launch_bounds__(256)
void conv5x5_wmma_k(const __bf16* __restrict__ y,    // [16][768][64][64]
                    const __bf16* __restrict__ wp,   // [25][768 cout][768 cin]
                    const float* __restrict__ conv_b,
                    float* __restrict__ hmap) {
  __shared__ __align__(16) __bf16 ldsA[2][ATILE];    // [kx][cout_local][cin_local]
  __shared__ __align__(16) __bf16 ldsB[2][BTILE];    // [wext(=w'+2)][cin_local]
  const int row  = blockIdx.x;
  const int mblk = blockIdx.y;
  const int img  = blockIdx.z;
  const int tid  = threadIdx.x;
  const int wave = tid >> 5;
  const int lane = tid & 31;
  const int l15  = lane & 15;
  const bool hih = lane >= 16;

  v8f acc[4] = {v8f{}, v8f{}, v8f{}, v8f{}};

  const int m_local = wave * 16 + l15;     // A-frag row within 128 tile
  const int ka0 = hih ? 8 : 0;             // A layout: lo lanes K0-7/16-23, hi K8-15/24-31
  const int kb0 = hih ? 8 : 0;             // B layout: lo lanes K0-7/16-23, hi K8-15/24-31

  // B staging indices
  const int b_k  = tid >> 3;               // cin_local 0..31
  const int b_w0 = (tid & 7) * 8;          // w base (interior)

  // stage(it, buf): issue async A loads + store B tile for iteration it
  auto stage = [&](int it, int buf) {
    const int ky   = it / 24;
    const int cc   = it % 24;
    const int cin0 = cc * 32;
    const int hp   = row + ky - 2;
    const bool rowok = (hp >= 0) && (hp < HW);
    // ---- A: 5 kx tiles of 128x32, 2560 b128 chunks, 10 per thread, async
    const unsigned aBase = (unsigned)(size_t)&ldsA[buf][0];
    #pragma unroll
    for (int i = 0; i < 10; ++i) {
      const int q   = tid + 256 * i;       // 0..2559
      const int kx  = q >> 9;              // 0..4
      const int rem = q & 511;
      const int co  = rem >> 2;            // 0..127
      const int kh  = (rem & 3) << 3;      // 0,8,16,24
      const __bf16* src = wp + ((long)(ky * 5 + kx) * C + (mblk * 128 + co)) * C + cin0 + kh;
      async_ld_b128(aBase + (unsigned)(kx * 4096 + co * 32 + kh) * 2u, src);
    }
    // ---- B: halo columns (w' = -2,-1,64,65) are always OOB -> zero
    if (tid < 128) {
      const int k = tid >> 2, p = tid & 3;
      const int wext = (p < 2) ? p : 64 + p;           // 0,1,66,67
      ldsB[buf][wext * 32 + k] = (__bf16)0.0f;
    }
    // ---- B interior: 32 cin x 64 w, transposed store [wext][cin]
    if (rowok) {
      const __bf16* srow = y + (((long)img * C + cin0 + b_k) * HW + hp) * HW + b_w0;
      uint4 d = *(const uint4*)srow;                   // 8 contiguous bf16
      const __bf16* dv = (const __bf16*)&d;
      #pragma unroll
      for (int j = 0; j < 8; ++j)
        ldsB[buf][(b_w0 + j + 2) * 32 + b_k] = dv[j];
    } else {
      #pragma unroll
      for (int j = 0; j < 8; ++j)
        ldsB[buf][(b_w0 + j + 2) * 32 + b_k] = (__bf16)0.0f;
    }
  };

  stage(0, 0);
  for (int it = 0; it < 120; ++it) {
    const int cur = it & 1;
    wait_async0();        // own wave's async copies (issued at stage(it)) complete
    __syncthreads();      // all waves' staging done; previous compute drained
    if (it + 1 < 120) stage(it + 1, cur ^ 1);
    // ---- compute: 5 kx x 4 N-subtiles = 20 WMMAs
    #pragma unroll
    for (int kx = 0; kx < 5; ++kx) {
      const __bf16* Ab = &ldsA[cur][kx * 4096 + m_local * 32];
      v16bf afrag = join16(*(const v8bf*)&Ab[ka0], *(const v8bf*)&Ab[ka0 + 16]);
      #pragma unroll
      for (int s = 0; s < 4; ++s) {
        const __bf16* Bb = &ldsB[cur][(s * 16 + l15 + kx) * 32 + kb0];
        v16bf bfrag = join16(*(const v8bf*)&Bb[0], *(const v8bf*)&Bb[16]);
        acc[s] = __builtin_amdgcn_wmma_f32_16x16x32_bf16(
            false, afrag, false, bfrag, (short)0, acc[s], false, false);
      }
    }
  }
  // ---- epilogue: C layout VGPR r -> M=r (lo lanes) / r+8 (hi lanes), N=lane&15
  #pragma unroll
  for (int r = 0; r < 8; ++r) {
    const int m    = wave * 16 + r + (hih ? 8 : 0);
    const int cout = mblk * 128 + m;
    const float bias = conv_b[cout];
    const long obase = (((long)img * C + cout) * HW + row) * HW;
    #pragma unroll
    for (int s = 0; s < 4; ++s)
      hmap[obase + s * 16 + l15] = acc[s][r] + bias;
  }
}

// -------------------------------------- spatial softmax + weighted pooling, per (n,c)
__global__ __launch_bounds__(256)
void softpool_k(const float* __restrict__ hmap, const __bf16* __restrict__ y,
                float* __restrict__ pooled) {
  const int nc = blockIdx.x;                   // 0..12287
  const long base = (long)nc * SPAT;
  __shared__ float r1[256], r2[256];
  const int tid = threadIdx.x;
  float mx = -1e30f;
  for (int i = tid; i < SPAT; i += 256) mx = fmaxf(mx, hmap[base + i]);
  r1[tid] = mx; __syncthreads();
  for (int off = 128; off > 0; off >>= 1) {
    if (tid < off) r1[tid] = fmaxf(r1[tid], r1[tid + off]);
    __syncthreads();
  }
  mx = r1[0]; __syncthreads();
  float se = 0.f, sxy = 0.f;
  for (int i = tid; i < SPAT; i += 256) {
    float e = __expf(hmap[base + i] - mx);
    se += e;
    sxy += e * (float)y[base + i];
  }
  r1[tid] = se; r2[tid] = sxy; __syncthreads();
  for (int off = 128; off > 0; off >>= 1) {
    if (tid < off) { r1[tid] += r1[tid + off]; r2[tid] += r2[tid + off]; }
    __syncthreads();
  }
  if (tid == 0) pooled[nc] = r2[0] / r1[0];
}

// --------------------------------------------- M=16 GEMM via one WMMA tile per wave:
// out[16][N] = X[16][K] @ W[N][K]^T (+bias)(+addsrc). fp32->bf16 on the fly.
__global__ __launch_bounds__(256)
void gemm16_wmma_k(const float* __restrict__ X, const float* __restrict__ W,
                   const float* __restrict__ bias, const float* __restrict__ addsrc,
                   float* __restrict__ out, int N, int K) {
  const int tid  = threadIdx.x;
  const int wave = tid >> 5, lane = tid & 31, l15 = lane & 15;
  const bool hih = lane >= 16;
  const int j0 = (blockIdx.x * 8 + wave) * 16;
  if (j0 >= N) return;                            // uniform per wave: EXEC stays full
  const int ka0 = hih ? 8 : 0;
  const int kb0 = hih ? 8 : 0;
  const int jcol = j0 + l15;
  v8f acc = {};
  for (int k0 = 0; k0 < K; k0 += 32) {
    v16bf a, b;
    const float* arow = X + l15 * K + k0;
    #pragma unroll
    for (int i = 0; i < 8; ++i) {
      a[i]     = (__bf16)arow[ka0 + i];
      a[i + 8] = (__bf16)arow[ka0 + 16 + i];
    }
    const float* brow = W + (long)jcol * K + k0 + kb0;
    #pragma unroll
    for (int i = 0; i < 8; ++i) { b[i] = (__bf16)brow[i]; b[i + 8] = (__bf16)brow[16 + i]; }
    acc = __builtin_amdgcn_wmma_f32_16x16x32_bf16(
        false, a, false, b, (short)0, acc, false, false);
  }
  #pragma unroll
  for (int r = 0; r < 8; ++r) {
    const int mm = r + (hih ? 8 : 0);
    float v = acc[r];
    if (bias)   v += bias[jcol];
    if (addsrc) v += addsrc[mm * N + jcol];
    out[mm * N + jcol] = v;
  }
}

// --------------------------------------------------- LayerNorm over last dim, 16 rows
__global__ __launch_bounds__(256)
void layernorm16_k(const float* __restrict__ x, const float* __restrict__ g,
                   const float* __restrict__ b, float* __restrict__ out, int N) {
  const int row = blockIdx.x, tid = threadIdx.x;
  __shared__ float r1[256], r2[256];
  float s = 0.f, s2 = 0.f;
  for (int j = tid; j < N; j += 256) { float v = x[row * N + j]; s += v; s2 += v * v; }
  r1[tid] = s; r2[tid] = s2; __syncthreads();
  for (int off = 128; off > 0; off >>= 1) {
    if (tid < off) { r1[tid] += r1[tid + off]; r2[tid] += r2[tid + off]; }
    __syncthreads();
  }
  const float mean = r1[0] / (float)N;
  const float var  = r2[0] / (float)N - mean * mean;
  const float inv  = rsqrtf(var + EPS);
  for (int j = tid; j < N; j += 256) {
    float v = x[row * N + j];
    out[row * N + j] = (v - mean) * inv * g[j] + b[j];
  }
}

// ------------------------------------- linear attention core, one block per (b,head)
// out = softmax_d(q) @ (softmax_n(k)^T v)  ==  (softq @ softk^T) @ v   (8x8 scores)
__global__ __launch_bounds__(128)
void attn_core_k(const float* __restrict__ q_, const float* __restrict__ k_,
                 const float* __restrict__ v_, float* __restrict__ out_) {
  const int b = blockIdx.x >> 3, head = blockIdx.x & 7;
  __shared__ float q[8][128], k[8][128], v[8][128], s[8][8], red[128];
  const int d = threadIdx.x;
  for (int n = 0; n < 8; ++n) {
    const long idx = ((long)(b * 8 + n)) * INNER + head * 128 + d;
    q[n][d] = q_[idx]; k[n][d] = k_[idx]; v[n][d] = v_[idx];
  }
  __syncthreads();
  for (int n = 0; n < 8; ++n) {           // softmax q over feature dim
    float val = q[n][d];
    red[d] = val; __syncthreads();
    for (int off = 64; off > 0; off >>= 1) { if (d < off) red[d] = fmaxf(red[d], red[d + off]); __syncthreads(); }
    const float mx = red[0]; __syncthreads();
    const float e = __expf(val - mx);
    red[d] = e; __syncthreads();
    for (int off = 64; off > 0; off >>= 1) { if (d < off) red[d] += red[d + off]; __syncthreads(); }
    const float sum = red[0]; __syncthreads();
    q[n][d] = e / sum; __syncthreads();
  }
  {                                        // softmax k over seq dim (per column)
    float mx = -1e30f;
    for (int n = 0; n < 8; ++n) mx = fmaxf(mx, k[n][d]);
    float e[8]; float sum = 0.f;
    for (int n = 0; n < 8; ++n) { e[n] = __expf(k[n][d] - mx); sum += e[n]; }
    for (int n = 0; n < 8; ++n) k[n][d] = e[n] / sum;
  }
  __syncthreads();
  for (int nm = 0; nm < 64; ++nm) {        // s[n][m] = sum_d q[n][d]*k[m][d]
    const int n = nm >> 3, m = nm & 7;
    red[d] = q[n][d] * k[m][d]; __syncthreads();
    for (int off = 64; off > 0; off >>= 1) { if (d < off) red[d] += red[d + off]; __syncthreads(); }
    if (d == 0) s[n][m] = red[0];
    __syncthreads();
  }
  for (int n = 0; n < 8; ++n) {            // out = s @ v
    float o = 0.f;
    #pragma unroll
    for (int m = 0; m < 8; ++m) o += s[n][m] * v[m][d];
    out_[((long)(b * 8 + n)) * INNER + head * 128 + d] = o;
  }
}

extern "C" void kernel_launch(void* const* d_in, const int* in_sizes, int n_in,
                              void* d_out, int out_size, void* d_ws, size_t ws_size,
                              hipStream_t stream) {
  (void)in_sizes; (void)n_in; (void)out_size; (void)ws_size;
  const float* x      = (const float*)d_in[0];
  const float* gn_g   = (const float*)d_in[1];
  const float* gn_b   = (const float*)d_in[2];
  const float* conv_w = (const float*)d_in[3];
  const float* conv_b = (const float*)d_in[4];
  const float* pin_w  = (const float*)d_in[5];
  const float* pin_b  = (const float*)d_in[6];
  const float* ln1_g  = (const float*)d_in[7];
  const float* ln1_b  = (const float*)d_in[8];
  const float* a1_wq  = (const float*)d_in[9];
  const float* a1_wk  = (const float*)d_in[10];
  const float* a1_wv  = (const float*)d_in[11];
  const float* a1_wo  = (const float*)d_in[12];
  const float* a1_bo  = (const float*)d_in[13];
  const float* ln2_g  = (const float*)d_in[14];
  const float* ln2_b  = (const float*)d_in[15];
  const float* a2_wq  = (const float*)d_in[16];
  const float* a2_wk  = (const float*)d_in[17];
  const float* a2_wv  = (const float*)d_in[18];
  const float* a2_wo  = (const float*)d_in[19];
  const float* a2_bo  = (const float*)d_in[20];
  // 21..26: ln3/ff* unused (reference discards _ff)
  const float* pout_w = (const float*)d_in[27];
  const float* pout_b = (const float*)d_in[28];

  char* ws = (char*)d_ws;
  size_t off = 0;
  auto carve = [&](size_t bytes) -> void* {
    void* p = ws + off;
    off += (bytes + 255) & ~(size_t)255;
    return p;
  };
  __bf16* y     = (__bf16*)carve((size_t)NIMG * C * SPAT * 2);     // 100.7 MB
  __bf16* wpk   = (__bf16*)carve((size_t)25 * C * C * 2);          //  29.5 MB
  float* hmap   = (float*) carve((size_t)NIMG * C * SPAT * 4);     // 201.3 MB
  float* stats  = (float*) carve(512 * 2 * 4);
  float* pooled = (float*) carve(16 * C * 4);
  float* H0 = (float*)carve(16 * INNER * 4);
  float* LN = (float*)carve(16 * INNER * 4);
  float* QB = (float*)carve(16 * INNER * 4);
  float* KB = (float*)carve(16 * INNER * 4);
  float* VB = (float*)carve(16 * INNER * 4);
  float* AO = (float*)carve(16 * INNER * 4);
  float* H1 = (float*)carve(16 * INNER * 4);
  float* H2 = (float*)carve(16 * INNER * 4);

  // 1) GroupNorm + SiLU -> bf16
  gn_stats_k<<<512, 256, 0, stream>>>(x, stats);
  gn_silu_k<<<49152, 256, 0, stream>>>(x, stats, gn_g, gn_b, y);
  // 2) conv weight repack + 5x5 conv (WMMA implicit GEMM, async-LDS pipelined)
  wpack_k<<<57600, 256, 0, stream>>>(conv_w, wpk);
  conv5x5_wmma_k<<<dim3(64, 6, 16), 256, 0, stream>>>(y, wpk, conv_b, hmap);
  // 3) spatial-softmax weighted pooling -> (16, 768)
  softpool_k<<<12288, 256, 0, stream>>>(hmap, y, pooled);
  // 4) pin projection: (16,768) -> (16,1024)
  gemm16_wmma_k<<<8, 256, 0, stream>>>(pooled, pin_w, pin_b, nullptr, H0, INNER, C);
  // 5) attention block 1
  layernorm16_k<<<16, 256, 0, stream>>>(H0, ln1_g, ln1_b, LN, INNER);
  gemm16_wmma_k<<<8, 256, 0, stream>>>(LN, a1_wq, nullptr, nullptr, QB, INNER, INNER);
  gemm16_wmma_k<<<8, 256, 0, stream>>>(LN, a1_wk, nullptr, nullptr, KB, INNER, INNER);
  gemm16_wmma_k<<<8, 256, 0, stream>>>(LN, a1_wv, nullptr, nullptr, VB, INNER, INNER);
  attn_core_k<<<16, 128, 0, stream>>>(QB, KB, VB, AO);
  gemm16_wmma_k<<<8, 256, 0, stream>>>(AO, a1_wo, a1_bo, H0, H1, INNER, INNER);
  // 6) attention block 2
  layernorm16_k<<<16, 256, 0, stream>>>(H1, ln2_g, ln2_b, LN, INNER);
  gemm16_wmma_k<<<8, 256, 0, stream>>>(LN, a2_wq, nullptr, nullptr, QB, INNER, INNER);
  gemm16_wmma_k<<<8, 256, 0, stream>>>(LN, a2_wk, nullptr, nullptr, KB, INNER, INNER);
  gemm16_wmma_k<<<8, 256, 0, stream>>>(LN, a2_wv, nullptr, nullptr, VB, INNER, INNER);
  attn_core_k<<<16, 128, 0, stream>>>(QB, KB, VB, AO);
  gemm16_wmma_k<<<8, 256, 0, stream>>>(AO, a2_wo, a2_bo, H1, H2, INNER, INNER);
  // 7) (_ff intentionally skipped: unused in reference)
  //    out = H2 @ pout_w^T + pout_b + residual(pooled)
  gemm16_wmma_k<<<6, 256, 0, stream>>>(H2, pout_w, pout_b, pooled, (float*)d_out, C, INNER);
}
